// LlamaMLP_21053929685151
// MI455X (gfx1250) — matrix operand order repaired and checked by reading, not compiled, not run on hardware
//
#include <hip/hip_runtime.h>

// ---------------------------------------------------------------------------
// LLaMA MLP + per-sequence LoRA for MI455X (gfx1250, wave32, WMMA).
//
// Roofline: 5.5e11 FLOPs vs 541MB fp32 weights -> compute-bound at bf16 WMMA
// rates. One-time fp32->bf16 conversion into workspace, then
// v_wmma_f32_16x16x32_bf16 GEMMs with f32 accumulation; bf16 weights per
// stage (<=180MB) stay L2-resident (192MB) across the 16 token-tiles.
// Tile movement uses CDNA5 GLOBAL_LOAD_ASYNC_TO_LDS_B128 (ASYNCcnt) in a
// steady-state double-buffered pipeline: the DMA of tile k+2 is issued right
// after tile k's compute, so one tile is always in flight during WMMA work.
// LoRA rank-16 corrections are folded in as extra WMMAs (K zero-padded to 32,
// lora_scale folded into the B-side), so everything hot is matrix-core work.
// ---------------------------------------------------------------------------

#define DEVINL __device__ __forceinline__

constexpr int BATCH = 4;
constexpr int SEQ   = 512;
constexpr int HID   = 4096;
constexpr int FFN   = 11008;
constexpr int RANK  = 16;
constexpr int NADP  = 16;      // num adapters R
constexpr float LSCALE = 0.5f; // lora scale
constexpr int TOK = BATCH * SEQ; // 2048 tokens

typedef __attribute__((ext_vector_type(16))) __bf16 v16bf;
typedef __attribute__((ext_vector_type(8)))  float  v8f;

union Frag { v16bf v; uint4 q[2]; };

DEVINL unsigned short f2bf(float f) {
  unsigned u = __builtin_bit_cast(unsigned, f);
  u += 0x7FFFu + ((u >> 16) & 1u);          // round-to-nearest-even
  return (unsigned short)(u >> 16);
}
DEVINL float bf2f(unsigned short h) {
  unsigned u = ((unsigned)h) << 16;
  return __builtin_bit_cast(float, u);
}

// --- CDNA5 async global->LDS copy (per-lane, 16B), tracked by ASYNCcnt ------
DEVINL void async_b128(unsigned lds_off, const void* gaddr) {
  asm volatile("global_load_async_to_lds_b128 %0, %1, off"
               :: "v"(lds_off), "v"(gaddr) : "memory");
}
template <int N> DEVINL void wait_asynccnt() {
  asm volatile("s_wait_asynccnt %0" :: "n"(N) : "memory");
}
DEVINL unsigned lds_off32(const void* p) {   // low 32 bits = LDS byte offset
  return (unsigned)(unsigned long long)p;
}

// ---------------------------------------------------------------------------
// Kernel 0: fp32 -> bf16 bulk conversion (vectorized, grid-stride).
// ---------------------------------------------------------------------------
__global__ void cvt_f32_to_bf16(const float* __restrict__ in,
                                unsigned short* __restrict__ out,
                                long long n4) {  // number of float4 chunks
  long long i = (long long)blockIdx.x * blockDim.x + threadIdx.x;
  long long stride = (long long)gridDim.x * blockDim.x;
  for (; i < n4; i += stride) {
    float4 f = ((const float4*)in)[i];
    ushort4 o;
    o.x = f2bf(f.x); o.y = f2bf(f.y); o.z = f2bf(f.z); o.w = f2bf(f.w);
    ((ushort4*)out)[i] = o;
  }
}

// ---------------------------------------------------------------------------
// Kernel 1: t_gate/t_up[m][r] = sum_h A[layer][idx[m]][r][h] * x[m][h]  (fp32)
// ---------------------------------------------------------------------------
__global__ __launch_bounds__(256)
void lora_down_gu(const float* __restrict__ x,
                  const float* __restrict__ Ag,
                  const float* __restrict__ Au,
                  const int* __restrict__ lora_idx,
                  const int* __restrict__ layer_idx,
                  float* __restrict__ tg,
                  float* __restrict__ tu) {
  int m = blockIdx.x;
  int b = m >> 9;              // 512 tokens per sequence
  int idx = lora_idx[b];
  long long rbase = ((long long)layer_idx[0] * NADP + idx) * RANK;
  int lane = threadIdx.x & 31, w = threadIdx.x >> 5;
  const float* xr = x + (long long)m * HID;
#pragma unroll
  for (int c = 0; c < 4; ++c) {
    int r = w + (c & 1) * 8;
    const float* A = ((c < 2) ? Ag : Au) + (rbase + r) * HID;
    float s = 0.f;
    for (int h = lane; h < HID; h += 32) s += xr[h] * A[h];
#pragma unroll
    for (int o = 16; o > 0; o >>= 1) s += __shfl_xor(s, o, 32);
    if (lane == 0) ((c < 2) ? tg : tu)[m * RANK + r] = s;
  }
}

// ---------------------------------------------------------------------------
// Kernel 3: t_down[m][r] = sum_f A_down[...][r][f] * act[m][f]  (act bf16)
// ---------------------------------------------------------------------------
__global__ __launch_bounds__(256)
void lora_down_d(const unsigned short* __restrict__ act,
                 const float* __restrict__ Ad,
                 const int* __restrict__ lora_idx,
                 const int* __restrict__ layer_idx,
                 float* __restrict__ td) {
  int m = blockIdx.x;
  int b = m >> 9;
  int idx = lora_idx[b];
  long long rbase = ((long long)layer_idx[0] * NADP + idx) * RANK;
  int lane = threadIdx.x & 31, w = threadIdx.x >> 5;
  const unsigned short* ar = act + (long long)m * FFN;
#pragma unroll
  for (int c = 0; c < 2; ++c) {
    int r = w + c * 8;
    const float* A = Ad + (rbase + r) * FFN;
    float s = 0.f;
    for (int f = lane; f < FFN; f += 32) s += bf2f(ar[f]) * A[f];
#pragma unroll
    for (int o = 16; o > 0; o >>= 1) s += __shfl_xor(s, o, 32);
    if (lane == 0) td[m * RANK + r] = s;
  }
}

// ---------------------------------------------------------------------------
// Fragment loaders (LDS row-major tiles, 32 bf16 = 64B per row).
// A (16x32, MxK): lane row = lane&15; lanes<16 hold K{0..7,16..23},
//                 lanes>=16 hold K{8..15,24..31}.
// B (32x16, KxN): lane col = lane&15; lanes<16 hold K{0..7,16..23},
//                 lanes>=16 hold K{8..15,24..31}.
// ---------------------------------------------------------------------------
DEVINL Frag load_a_frag(const char* tile, int row, int lane) {
  Frag a;
  const char* base = tile + row * 64 + ((lane & 16) ? 16 : 0);
  a.q[0] = *(const uint4*)(base);
  a.q[1] = *(const uint4*)(base + 32);
  return a;
}
DEVINL Frag load_b_frag(const char* tile, int col, int lane) {
  Frag b;
  const char* base = tile + col * 64 + ((lane & 16) ? 16 : 0);
  b.q[0] = *(const uint4*)(base);
  b.q[1] = *(const uint4*)(base + 32);
  return b;
}
DEVINL v8f wmma_bf16(Frag a, Frag b, v8f c) {
  return __builtin_amdgcn_wmma_f32_16x16x32_bf16(
      false, a.v, false, b.v, (short)0, c, false, false);
}

// ---------------------------------------------------------------------------
// Kernel 2: fused gate/up GEMM + LoRA-B (as WMMA) + SwiGLU -> act (bf16).
// Block tile: 128 tokens x 64 FFN cols. Waves 0-3: gate, waves 4-7: up.
// Double-buffered LDS, async global->LDS DMA (steady-state pipeline):
//   buf0 @0, buf1 @16384; within a buffer: xs[128][32]bf16 (8K),
//   wg[64][32] (+8192, 4K), wu[64][32] (+12288, 4K).
//   epilogue : gate_ex[128][64]f32 @0 (32K), tg_bf @32768 (8K),
//              tu_bf @40960 (8K), bg_bf @49152 (4K), bu_bf @53248 (4K)
// total 57344 bytes.
// ---------------------------------------------------------------------------
__global__ __launch_bounds__(256)
void gemm_gateup(const unsigned short* __restrict__ xb,  // [TOK][HID] bf16
                 const unsigned short* __restrict__ wg,  // [FFN][HID] bf16
                 const unsigned short* __restrict__ wu,  // [FFN][HID] bf16
                 const float* __restrict__ Bg,           // [L][R][FFN][RANK]
                 const float* __restrict__ Bu,
                 const float* __restrict__ tg,           // [TOK][RANK]
                 const float* __restrict__ tu,
                 const int* __restrict__ lora_idx,
                 const int* __restrict__ layer_idx,
                 unsigned short* __restrict__ actb) {    // [TOK][FFN] bf16
  extern __shared__ __align__(16) char smem[];

  const int F0 = blockIdx.x * 64;
  const int M0 = blockIdx.y * 128;
  const int tid = threadIdx.x, lane = tid & 31, w = tid >> 5;
  const bool isGate = (w < 4);
  const int wr = w & 3;                  // wave's 32-row group

  v8f zero = {0.f, 0.f, 0.f, 0.f, 0.f, 0.f, 0.f, 0.f};
  v8f acc[2][4];
#pragma unroll
  for (int mi = 0; mi < 2; ++mi)
#pragma unroll
    for (int n = 0; n < 4; ++n) acc[mi][n] = zero;

  // per-thread async-copy assignments (4 x b128 per thread per K-step)
  const unsigned lo = lds_off32(smem);
  const int ch0 = tid, ch1 = tid + 256;          // xs chunks 0..511
  const unsigned xs_o0 = lo + ch0 * 16;
  const unsigned xs_o1 = lo + ch1 * 16;
  const unsigned wg_o  = lo + 8192  + tid * 16;
  const unsigned wu_o  = lo + 12288 + tid * 16;
  const unsigned short* gx0 = xb + (long long)(M0 + (ch0 >> 2)) * HID + (ch0 & 3) * 8;
  const unsigned short* gx1 = xb + (long long)(M0 + (ch1 >> 2)) * HID + (ch1 & 3) * 8;
  const unsigned short* gwg = wg + (long long)(F0 + (tid >> 2)) * HID + (tid & 3) * 8;
  const unsigned short* gwu = wu + (long long)(F0 + (tid >> 2)) * HID + (tid & 3) * 8;

  auto prefetch = [&](int kt, unsigned bo) {
    const long long k0 = (long long)kt * 32;
    async_b128(xs_o0 + bo, gx0 + k0);
    async_b128(xs_o1 + bo, gx1 + k0);
    async_b128(wg_o + bo, gwg + k0);
    async_b128(wu_o + bo, gwu + k0);
  };

  const int KT = HID / 32;   // 128
  prefetch(0, 0u);
  prefetch(1, 16384u);
  unsigned bo = 0;
#pragma clang loop unroll(disable)
  for (int kt = 0; kt < KT; ++kt) {
    wait_asynccnt<4>();      // current buffer's 4 copies retired, next in flight
    __syncthreads();

    const char* bx = smem + bo;
    Frag a0 = load_a_frag(bx, wr * 32 + (lane & 15), lane);
    Frag a1 = load_a_frag(bx, wr * 32 + 16 + (lane & 15), lane);
    const char* wsel = bx + (isGate ? 8192 : 12288);
#pragma unroll
    for (int n = 0; n < 4; ++n) {
      Frag bf = load_b_frag(wsel, n * 16 + (lane & 15), lane);
      acc[0][n] = wmma_bf16(a0, bf, acc[0][n]);
      acc[1][n] = wmma_bf16(a1, bf, acc[1][n]);
    }
    __syncthreads();

    int kn = kt + 2;
    if (kn >= KT) kn -= KT;   // tail: wrapped (redundant) prefetch, retired below
    prefetch(kn, bo);
    bo ^= 16384u;
  }
  wait_asynccnt<0>();         // retire stray tail prefetches before LDS reuse
  __syncthreads();

  // ---- epilogue: LoRA-B correction as one extra WMMA per C-tile ----
  const int b = M0 >> 9;
  const int idx = lora_idx[b];
  const long long lbase = (long long)layer_idx[0] * NADP + idx;

  float* gate_ex = (float*)(smem);                          // 128*64 f32
  unsigned short* tg_bf = (unsigned short*)(smem + 32768);  // 128*32
  unsigned short* tu_bf = (unsigned short*)(smem + 40960);  // 128*32
  unsigned short* bg_bf = (unsigned short*)(smem + 49152);  // 64*32
  unsigned short* bu_bf = (unsigned short*)(smem + 53248);  // 64*32

  for (int i = tid; i < 128 * 32; i += 256) {
    int row = i >> 5, r = i & 31;
    tg_bf[i] = (r < RANK) ? f2bf(tg[(M0 + row) * RANK + r]) : (unsigned short)0;
    tu_bf[i] = (r < RANK) ? f2bf(tu[(M0 + row) * RANK + r]) : (unsigned short)0;
  }
  for (int i = tid; i < 64 * 32; i += 256) {
    int f = i >> 5, r = i & 31;
    bg_bf[i] = (r < RANK) ? f2bf(LSCALE * Bg[(lbase * FFN + F0 + f) * RANK + r])
                          : (unsigned short)0;
    bu_bf[i] = (r < RANK) ? f2bf(LSCALE * Bu[(lbase * FFN + F0 + f) * RANK + r])
                          : (unsigned short)0;
  }
  __syncthreads();

  {
    const char* tsel = isGate ? (const char*)tg_bf : (const char*)tu_bf;
    const char* bsel = isGate ? (const char*)bg_bf : (const char*)bu_bf;
    Frag a0 = load_a_frag(tsel, wr * 32 + (lane & 15), lane);
    Frag a1 = load_a_frag(tsel, wr * 32 + 16 + (lane & 15), lane);
#pragma unroll
    for (int n = 0; n < 4; ++n) {
      Frag bf = load_b_frag(bsel, n * 16 + (lane & 15), lane);
      acc[0][n] = wmma_bf16(a0, bf, acc[0][n]);
      acc[1][n] = wmma_bf16(a1, bf, acc[1][n]);
    }
  }
  __syncthreads();

  // gate waves: silu(gate) -> LDS exchange buffer
  if (isGate) {
#pragma unroll
    for (int mi = 0; mi < 2; ++mi)
#pragma unroll
      for (int n = 0; n < 4; ++n)
#pragma unroll
        for (int v = 0; v < 8; ++v) {
          int row = wr * 32 + mi * 16 + v + ((lane & 16) ? 8 : 0);
          int fl = n * 16 + (lane & 15);
          float g = acc[mi][n][v];
          float sig = 1.f / (1.f + __expf(-g));
          gate_ex[row * 64 + fl] = g * sig;
        }
  }
  __syncthreads();

  // up waves: act = silu(gate) * up, store bf16
  if (!isGate) {
#pragma unroll
    for (int mi = 0; mi < 2; ++mi)
#pragma unroll
      for (int n = 0; n < 4; ++n)
#pragma unroll
        for (int v = 0; v < 8; ++v) {
          int row = wr * 32 + mi * 16 + v + ((lane & 16) ? 8 : 0);
          int fl = n * 16 + (lane & 15);
          float u = acc[mi][n][v];
          float a = gate_ex[row * 64 + fl] * u;
          actb[(long long)(M0 + row) * FFN + F0 + fl] = f2bf(a);
        }
  }
}

// ---------------------------------------------------------------------------
// Kernel 4: down GEMM + LoRA-B WMMA epilogue -> fp32 output.
// Block tile: 128 tokens x 128 HID cols; 8 waves in 4x2 layout (32x64 each).
// Double-buffered LDS + async DMA (steady-state): buf0 @0, buf1 @16384;
// within a buffer: as[128][32] (8K), ws[128][32] (+8192, 8K).
// Epilogue reuses @0..16K. total 32768 bytes.
// ---------------------------------------------------------------------------
__global__ __launch_bounds__(256)
void gemm_down(const unsigned short* __restrict__ actb,  // [TOK][FFN] bf16
               const unsigned short* __restrict__ wd,    // [HID][FFN] bf16
               const float* __restrict__ Bd,             // [L][R][HID][RANK]
               const float* __restrict__ td,             // [TOK][RANK]
               const int* __restrict__ lora_idx,
               const int* __restrict__ layer_idx,
               float* __restrict__ out) {                // [TOK][HID] f32
  extern __shared__ __align__(16) char smem[];

  const int H0 = blockIdx.x * 128;
  const int M0 = blockIdx.y * 128;
  const int tid = threadIdx.x, lane = tid & 31, w = tid >> 5;
  const int wr = w & 3;       // 32-row group
  const int wc = w >> 2;      // 64-col group

  v8f zero = {0.f, 0.f, 0.f, 0.f, 0.f, 0.f, 0.f, 0.f};
  v8f acc[2][4];
#pragma unroll
  for (int mi = 0; mi < 2; ++mi)
#pragma unroll
    for (int n = 0; n < 4; ++n) acc[mi][n] = zero;

  const unsigned lo = lds_off32(smem);
  const int ch0 = tid, ch1 = tid + 256;
  const unsigned as_o0 = lo + ch0 * 16;
  const unsigned as_o1 = lo + ch1 * 16;
  const unsigned ws_o0 = lo + 8192 + ch0 * 16;
  const unsigned ws_o1 = lo + 8192 + ch1 * 16;
  const unsigned short* ga0 = actb + (long long)(M0 + (ch0 >> 2)) * FFN + (ch0 & 3) * 8;
  const unsigned short* ga1 = actb + (long long)(M0 + (ch1 >> 2)) * FFN + (ch1 & 3) * 8;
  const unsigned short* gw0 = wd + (long long)(H0 + (ch0 >> 2)) * FFN + (ch0 & 3) * 8;
  const unsigned short* gw1 = wd + (long long)(H0 + (ch1 >> 2)) * FFN + (ch1 & 3) * 8;

  auto prefetch = [&](int kt, unsigned bo) {
    const long long k0 = (long long)kt * 32;
    async_b128(as_o0 + bo, ga0 + k0);
    async_b128(as_o1 + bo, ga1 + k0);
    async_b128(ws_o0 + bo, gw0 + k0);
    async_b128(ws_o1 + bo, gw1 + k0);
  };

  const int KT = FFN / 32;   // 344
  prefetch(0, 0u);
  prefetch(1, 16384u);
  unsigned bo = 0;
#pragma clang loop unroll(disable)
  for (int kt = 0; kt < KT; ++kt) {
    wait_asynccnt<4>();
    __syncthreads();

    const char* bx = smem + bo;
    const char* bw = bx + 8192;
    Frag a0 = load_a_frag(bx, wr * 32 + (lane & 15), lane);
    Frag a1 = load_a_frag(bx, wr * 32 + 16 + (lane & 15), lane);
#pragma unroll
    for (int n = 0; n < 4; ++n) {
      Frag bf = load_b_frag(bw, wc * 64 + n * 16 + (lane & 15), lane);
      acc[0][n] = wmma_bf16(a0, bf, acc[0][n]);
      acc[1][n] = wmma_bf16(a1, bf, acc[1][n]);
    }
    __syncthreads();

    int kn = kt + 2;
    if (kn >= KT) kn -= KT;   // tail: wrapped (redundant) prefetch
    prefetch(kn, bo);
    bo ^= 16384u;
  }
  wait_asynccnt<0>();
  __syncthreads();

  // ---- LoRA-B epilogue as WMMA ----
  const int b = M0 >> 9;
  const int idx = lora_idx[b];
  const long long lbase = (long long)layer_idx[0] * NADP + idx;

  unsigned short* td_bf = (unsigned short*)(smem);         // 128*32
  unsigned short* bd_bf = (unsigned short*)(smem + 8192);  // 128*32

  for (int i = tid; i < 128 * 32; i += 256) {
    int row = i >> 5, r = i & 31;
    td_bf[i] = (r < RANK) ? f2bf(td[(M0 + row) * RANK + r]) : (unsigned short)0;
    bd_bf[i] = (r < RANK) ? f2bf(LSCALE * Bd[(lbase * HID + H0 + row) * RANK + r])
                          : (unsigned short)0;
  }
  __syncthreads();

  {
    Frag a0 = load_a_frag((const char*)td_bf, wr * 32 + (lane & 15), lane);
    Frag a1 = load_a_frag((const char*)td_bf, wr * 32 + 16 + (lane & 15), lane);
#pragma unroll
    for (int n = 0; n < 4; ++n) {
      Frag bf = load_b_frag((const char*)bd_bf, wc * 64 + n * 16 + (lane & 15), lane);
      acc[0][n] = wmma_bf16(a0, bf, acc[0][n]);
      acc[1][n] = wmma_bf16(a1, bf, acc[1][n]);
    }
  }

#pragma unroll
  for (int mi = 0; mi < 2; ++mi)
#pragma unroll
    for (int n = 0; n < 4; ++n)
#pragma unroll
      for (int v = 0; v < 8; ++v) {
        int row = wr * 32 + mi * 16 + v + ((lane & 16) ? 8 : 0);
        int hl = wc * 64 + n * 16 + (lane & 15);
        out[(long long)(M0 + row) * HID + H0 + hl] = acc[mi][n][v];
      }
}

// ---------------------------------------------------------------------------
// Host launcher.
// Input order: x, W_gate, W_up, W_down, A_gate, B_gate, A_up, B_up,
//              A_down, B_down, lora_indices, layer_idx
// ---------------------------------------------------------------------------
extern "C" void kernel_launch(void* const* d_in, const int* in_sizes, int n_in,
                              void* d_out, int out_size, void* d_ws, size_t ws_size,
                              hipStream_t stream) {
  const float* x      = (const float*)d_in[0];
  const float* W_gate = (const float*)d_in[1];
  const float* W_up   = (const float*)d_in[2];
  const float* W_down = (const float*)d_in[3];
  const float* A_gate = (const float*)d_in[4];
  const float* B_gate = (const float*)d_in[5];
  const float* A_up   = (const float*)d_in[6];
  const float* B_up   = (const float*)d_in[7];
  const float* A_down = (const float*)d_in[8];
  const float* B_down = (const float*)d_in[9];
  const int* lora_idx = (const int*)d_in[10];
  const int* layer_ix = (const int*)d_in[11];
  float* out = (float*)d_out;

  // workspace layout
  const long long WH = (long long)FFN * HID;     // 45,088,768 elems
  char* ws = (char*)d_ws;
  unsigned short* Wg_bf  = (unsigned short*)(ws);
  unsigned short* Wu_bf  = (unsigned short*)(ws + WH * 2);
  unsigned short* Wd_bf  = (unsigned short*)(ws + WH * 4);
  unsigned short* x_bf   = (unsigned short*)(ws + WH * 6);
  unsigned short* act_bf = (unsigned short*)(ws + WH * 6 + (long long)TOK * HID * 2);
  char* tail = ws + WH * 6 + (long long)TOK * HID * 2 + (long long)TOK * FFN * 2;
  float* t_gate = (float*)(tail);
  float* t_up   = (float*)(tail + TOK * RANK * 4);
  float* t_down = (float*)(tail + 2 * TOK * RANK * 4);

  // 0) fp32 -> bf16 conversions
  cvt_f32_to_bf16<<<2048, 256, 0, stream>>>(W_gate, Wg_bf, WH / 4);
  cvt_f32_to_bf16<<<2048, 256, 0, stream>>>(W_up,   Wu_bf, WH / 4);
  cvt_f32_to_bf16<<<2048, 256, 0, stream>>>(W_down, Wd_bf, WH / 4);
  cvt_f32_to_bf16<<<512,  256, 0, stream>>>(x, x_bf, (long long)TOK * HID / 4);

  // 1) rank projections for gate/up (fp32 precision)
  lora_down_gu<<<TOK, 256, 0, stream>>>(x, A_gate, A_up, lora_idx, layer_ix,
                                        t_gate, t_up);

  // 2) fused gate/up GEMM + LoRA + SwiGLU -> act (bf16)
  {
    dim3 grid(FFN / 64, TOK / 128);   // (172, 16)
    gemm_gateup<<<grid, 256, 57344, stream>>>(x_bf, Wg_bf, Wu_bf,
                                              B_gate, B_up, t_gate, t_up,
                                              lora_idx, layer_ix, act_bf);
  }

  // 3) rank projection for down
  lora_down_d<<<TOK, 256, 0, stream>>>(act_bf, A_down, lora_idx, layer_ix,
                                       t_down);

  // 4) down GEMM + LoRA -> fp32 out
  {
    dim3 grid(HID / 128, TOK / 128);  // (32, 16)
    gemm_down<<<grid, 256, 32768, stream>>>(act_bf, Wd_bf, B_down, t_down,
                                            lora_idx, layer_ix, out);
  }
  (void)in_sizes; (void)n_in; (void)out_size; (void)ws_size;
}
